// TemporalHead_9397388443687
// MI455X (gfx1250) — matrix operand order repaired and checked by reading, not compiled, not run on hardware
//
#include <hip/hip_runtime.h>
#include <hip/hip_bf16.h>

#define T_SEQ 4096
#define NBATCH 4
#define C_DIM 1024
#define H_DIM 128
#define KPAD 136            // padded LDS row stride in ushorts (272 B -> bank-staggered)

typedef __attribute__((ext_vector_type(16))) __bf16 v16bf;
typedef __attribute__((ext_vector_type(8)))  float  v8f;

union BV { v16bf v; unsigned short u[16]; uint4 q[2]; };

__device__ __forceinline__ unsigned short f2bf(float f) {
    unsigned u = __builtin_bit_cast(unsigned, f);
    u += 0x7FFFu + ((u >> 16) & 1u);   // round-to-nearest-even
    return (unsigned short)(u >> 16);
}

#if __has_builtin(__builtin_amdgcn_cvt_pk_bf16_f32)
typedef __attribute__((ext_vector_type(2))) __bf16 v2bf;
__device__ __forceinline__ unsigned pk2bf(float a, float b) {
    v2bf r = __builtin_amdgcn_cvt_pk_bf16_f32(a, b);
    return __builtin_bit_cast(unsigned, r);
}
#else
__device__ __forceinline__ unsigned pk2bf(float a, float b) {
    return (unsigned)f2bf(a) | ((unsigned)f2bf(b) << 16);
}
#endif

// ---------------------------------------------------------------------------
// Pre-pass: f32 -> bf16 bulk conversion (bandwidth bound; 8 elems/thread)
// ---------------------------------------------------------------------------
__global__ __launch_bounds__(256)
void TemporalHead_cvt_kernel(const float* __restrict__ in,
                             unsigned short* __restrict__ out, int n8) {
    const int i = blockIdx.x * 256 + threadIdx.x;
    if (i >= n8) return;
    const float4* src = (const float4*)in + (size_t)i * 2;
    const float4 a = src[0];
    const float4 b = src[1];
    uint4 r;
    r.x = pk2bf(a.x, a.y);
    r.y = pk2bf(a.z, a.w);
    r.z = pk2bf(b.x, b.y);
    r.w = pk2bf(b.z, b.w);
    ((uint4*)out)[i] = r;
}

// ---------------------------------------------------------------------------
// Kernel 1: QKV projection from pre-converted bf16 X / W.
// One wave -> one 16x16 tile of Q, K, V; 2-stage software pipeline so the
// 8 b128 loads of iteration i+1 are in flight during iteration i's WMMAs.
// ---------------------------------------------------------------------------
__global__ __launch_bounds__(32)
void TemporalHead_qkv_kernel(const unsigned short* __restrict__ xb,
                             const unsigned short* __restrict__ wq,
                             const unsigned short* __restrict__ wk,
                             const unsigned short* __restrict__ wv,
                             unsigned short* __restrict__ qo,
                             unsigned short* __restrict__ ko,
                             unsigned short* __restrict__ vo) {
    const int lane = threadIdx.x & 31;
    const int hi   = lane >> 4;
    const int ln   = lane & 15;
    const int row0 = blockIdx.x * 16;
    const int col0 = blockIdx.y * 16;

    v8f aq = {}; v8f ak = {}; v8f av = {};
    const unsigned short* xrow = xb + (size_t)(row0 + ln) * C_DIM;
    const unsigned short* wqr  = wq + (size_t)(col0 + ln) * C_DIM;
    const unsigned short* wkr  = wk + (size_t)(col0 + ln) * C_DIM;
    const unsigned short* wvr  = wv + (size_t)(col0 + ln) * C_DIM;

    BV a[2], bq[2], bk[2], bv[2];
    auto LOAD = [&](int s, int kk) {
        // A 16x32 bf16: lane half 'hi' holds K = hi*16 + 0..15 (contiguous)
        a[s].q[0]  = *(const uint4*)(xrow + kk + (hi << 4));
        a[s].q[1]  = *(const uint4*)(xrow + kk + (hi << 4) + 8);
        // B 32x16 bf16: lane half 'hi' holds K = hi*16 + 0..15 (contiguous)
        bq[s].q[0] = *(const uint4*)(wqr + kk + (hi << 4));
        bq[s].q[1] = *(const uint4*)(wqr + kk + (hi << 4) + 8);
        bk[s].q[0] = *(const uint4*)(wkr + kk + (hi << 4));
        bk[s].q[1] = *(const uint4*)(wkr + kk + (hi << 4) + 8);
        bv[s].q[0] = *(const uint4*)(wvr + kk + (hi << 4));
        bv[s].q[1] = *(const uint4*)(wvr + kk + (hi << 4) + 8);
    };

    LOAD(0, 0);
#pragma unroll 4
    for (int it = 0; it < C_DIM / 32; ++it) {
        const int cur = it & 1;
        if (it + 1 < C_DIM / 32) LOAD(cur ^ 1, (it + 1) * 32);
        aq = __builtin_amdgcn_wmma_f32_16x16x32_bf16(false, a[cur].v, false, bq[cur].v, (short)0, aq, false, false);
        ak = __builtin_amdgcn_wmma_f32_16x16x32_bf16(false, a[cur].v, false, bk[cur].v, (short)0, ak, false, false);
        av = __builtin_amdgcn_wmma_f32_16x16x32_bf16(false, a[cur].v, false, bv[cur].v, (short)0, av, false, false);
    }

    const float qscale = 0.08838834764831845f;  // 1/sqrt(128)
#pragma unroll
    for (int r = 0; r < 8; ++r) {
        const int row = row0 + r + (hi << 3);          // C/D: M = r + 8*hi
        const size_t off = (size_t)row * H_DIM + col0 + ln;
        qo[off] = f2bf(aq[r] * qscale);
        ko[off] = f2bf(ak[r]);
        vo[off] = f2bf(av[r]);
    }
}

// ---------------------------------------------------------------------------
// One 32-key flash-attention block (MASK only instantiated for diagonal tiles)
// ---------------------------------------------------------------------------
template <bool MASK>
__device__ __forceinline__ void attn_block(const unsigned short* __restrict__ Ktb,
                                           const unsigned short* __restrict__ Vtb,
                                           unsigned short* __restrict__ Pt,
                                           const BV (&Aq)[4], v8f (&O)[8],
                                           float (&rowmax)[8], float (&rowsum)[8],
                                           int q0, int ks, int hi, int ln) {
    // --- S = Q @ K^T : two 16x16 tiles -------------------------------------
    v8f S[2];
#pragma unroll
    for (int s = 0; s < 2; ++s) {
        v8f acc = {};
#pragma unroll
        for (int c = 0; c < 4; ++c) {
            BV bkv;
            const unsigned short* kr = Ktb + (s * 16 + ln) * KPAD + c * 32 + (hi << 4);
            bkv.q[0] = *(const uint4*)(kr);
            bkv.q[1] = *(const uint4*)(kr + 8);
            acc = __builtin_amdgcn_wmma_f32_16x16x32_bf16(false, Aq[c].v, false, bkv.v,
                                                          (short)0, acc, false, false);
        }
        S[s] = acc;
    }

    // --- causal mask + online softmax --------------------------------------
#pragma unroll
    for (int r = 0; r < 8; ++r) {
        const int qg = q0 + r + (hi << 3);
        if (MASK) {
#pragma unroll
            for (int s = 0; s < 2; ++s) {
                const int kg = ks + s * 16 + ln;
                if (kg > qg) S[s][r] = -1e30f;
            }
        }
        float v = fmaxf(S[0][r], S[1][r]);
        v = fmaxf(v, __shfl_xor(v, 8, 16));
        v = fmaxf(v, __shfl_xor(v, 4, 16));
        v = fmaxf(v, __shfl_xor(v, 2, 16));
        v = fmaxf(v, __shfl_xor(v, 1, 16));
        const float mnew  = fmaxf(rowmax[r], v);
        const float scale = __expf(rowmax[r] - mnew);
        const float p0 = __expf(S[0][r] - mnew);
        const float p1 = __expf(S[1][r] - mnew);
        float ps = p0 + p1;
        ps += __shfl_xor(ps, 8, 16);
        ps += __shfl_xor(ps, 4, 16);
        ps += __shfl_xor(ps, 2, 16);
        ps += __shfl_xor(ps, 1, 16);
        rowsum[r] = rowsum[r] * scale + ps;
        rowmax[r] = mnew;
#pragma unroll
        for (int h = 0; h < 8; ++h) O[h][r] *= scale;
        Pt[(r + (hi << 3)) * 32 + ln]      = f2bf(p0);   // C-layout -> LDS
        Pt[(r + (hi << 3)) * 32 + 16 + ln] = f2bf(p1);
    }
    __syncthreads();

    // --- P as A-operand (16 queries x 32 keys) -----------------------------
    BV Ap;
    Ap.q[0] = *(const uint4*)(Pt + ln * 32 + (hi << 4));
    Ap.q[1] = *(const uint4*)(Pt + ln * 32 + (hi << 4) + 8);

    // --- O += P @ V over 8 head chunks -------------------------------------
#pragma unroll
    for (int h = 0; h < 8; ++h) {
        BV bvv;
#pragma unroll
        for (int j = 0; j < 16; ++j)
            bvv.u[j] = Vtb[((hi << 4) + j) * KPAD + h * 16 + ln];
        O[h] = __builtin_amdgcn_wmma_f32_16x16x32_bf16(false, Ap.v, false, bvv.v,
                                                       (short)0, O[h], false, false);
    }
    __syncthreads();
}

// ---------------------------------------------------------------------------
// Kernel 2: causal flash attention; K/V double-buffered in LDS, filled by
// async-to-LDS DMA overlapped with WMMA on the other buffer.
// ---------------------------------------------------------------------------
__global__ __launch_bounds__(32)
void TemporalHead_attn_kernel(const unsigned short* __restrict__ qb,
                              const unsigned short* __restrict__ kb,
                              const unsigned short* __restrict__ vb,
                              float* __restrict__ out) {
    __shared__ unsigned short Kt[2][32 * KPAD];
    __shared__ unsigned short Vt[2][32 * KPAD];
    __shared__ unsigned short Pt[16 * 32];

    const int lane = threadIdx.x & 31;
    const int hi   = lane >> 4;
    const int ln   = lane & 15;
    const int q0   = blockIdx.x * 16;
    const size_t base = (size_t)blockIdx.y * T_SEQ * H_DIM;

    // Q tile as 4 A-operands (head chunks of 32), resident for whole kernel
    BV Aq[4];
    {
        const unsigned short* qrow = qb + base + (size_t)(q0 + ln) * H_DIM;
#pragma unroll
        for (int c = 0; c < 4; ++c) {
            Aq[c].q[0] = *(const uint4*)(qrow + c * 32 + (hi << 4));
            Aq[c].q[1] = *(const uint4*)(qrow + c * 32 + (hi << 4) + 8);
        }
    }

    v8f O[8];
#pragma unroll
    for (int h = 0; h < 8; ++h) O[h] = (v8f){0.f,0.f,0.f,0.f,0.f,0.f,0.f,0.f};
    float rowmax[8], rowsum[8];
#pragma unroll
    for (int r = 0; r < 8; ++r) { rowmax[r] = -1e30f; rowsum[r] = 0.f; }

    const unsigned kt_lds[2] = {
        (unsigned)(size_t)(&Kt[0][0]) + (unsigned)(lane * KPAD * 2),
        (unsigned)(size_t)(&Kt[1][0]) + (unsigned)(lane * KPAD * 2) };
    const unsigned vt_lds[2] = {
        (unsigned)(size_t)(&Vt[0][0]) + (unsigned)(lane * KPAD * 2),
        (unsigned)(size_t)(&Vt[1][0]) + (unsigned)(lane * KPAD * 2) };

    // issue one 32-row K/V tile (32 async b128 per lane-row) into buffer b
    auto stage = [&](int b, int ks) {
        const unsigned short* kptr = kb + base + (size_t)(ks + lane) * H_DIM;
        const unsigned short* vptr = vb + base + (size_t)(ks + lane) * H_DIM;
        const unsigned kd = kt_lds[b], vd = vt_lds[b];
        // DScnt and ASYNCcnt are mutually unordered: make sure all LDS reads of
        // this buffer (2 iterations ago) have drained before the DMA overwrites.
        asm volatile("s_wait_dscnt 0x0" ::: "memory");
#pragma unroll
        for (int i = 0; i < 16; ++i) {
            asm volatile("global_load_async_to_lds_b128 %0, %1, off offset:%2"
                         :: "v"(kd), "v"(kptr), "i"(i * 16) : "memory");
            asm volatile("global_load_async_to_lds_b128 %0, %1, off offset:%2"
                         :: "v"(vd), "v"(vptr), "i"(i * 16) : "memory");
        }
    };

    const int nb     = (q0 + 47) >> 5;   // 32-key blocks covering keys 0..q0+15
    const int nbFull = q0 >> 5;          // blocks strictly below the diagonal

    stage(0, 0);
    for (int blk = 0; blk < nb; ++blk) {
        const int cur = blk & 1;
        const int ks  = blk * 32;

        if (blk + 1 < nb) {
            stage(cur ^ 1, ks + 32);     // DMA next tile into the other buffer
            // async loads complete in order: <=32 outstanding => current done
            asm volatile("s_wait_asynccnt 0x20" ::: "memory");
        } else {
            asm volatile("s_wait_asynccnt 0x0" ::: "memory");
        }
        if (ks + 64 < T_SEQ) {           // warm L2 two tiles ahead
            __builtin_prefetch(kb + base + (size_t)(ks + 64 + lane) * H_DIM, 0, 0);
            __builtin_prefetch(vb + base + (size_t)(ks + 64 + lane) * H_DIM, 0, 0);
        }

        if (blk < nbFull)
            attn_block<false>(Kt[cur], Vt[cur], Pt, Aq, O, rowmax, rowsum, q0, ks, hi, ln);
        else
            attn_block<true >(Kt[cur], Vt[cur], Pt, Aq, O, rowmax, rowsum, q0, ks, hi, ln);
    }

    // --- epilogue: normalize, store fp32 -----------------------------------
#pragma unroll
    for (int r = 0; r < 8; ++r) {
        const float inv = 1.0f / rowsum[r];
        const size_t off = base + (size_t)(q0 + r + (hi << 3)) * H_DIM + ln;
#pragma unroll
        for (int h = 0; h < 8; ++h)
            out[off + h * 16] = O[h][r] * inv;
    }
}

extern "C" void kernel_launch(void* const* d_in, const int* in_sizes, int n_in,
                              void* d_out, int out_size, void* d_ws, size_t ws_size,
                              hipStream_t stream) {
    (void)in_sizes; (void)n_in; (void)out_size; (void)ws_size;
    const float* x  = (const float*)d_in[0];
    const float* Wq = (const float*)d_in[1];
    const float* Wk = (const float*)d_in[2];
    const float* Wv = (const float*)d_in[3];

    const size_t XN  = (size_t)NBATCH * T_SEQ * C_DIM;   // 16M elems
    const size_t WN  = (size_t)H_DIM * C_DIM;            // 128K elems
    const size_t tok = (size_t)NBATCH * T_SEQ * H_DIM;   // 2M elems

    unsigned short* xb = (unsigned short*)d_ws;          // 32 MB
    unsigned short* wq = xb + XN;                        // 256 KB
    unsigned short* wk = wq + WN;
    unsigned short* wv = wk + WN;
    unsigned short* qb = wv + WN;                        // 4 MB
    unsigned short* kbuf = qb + tok;                     // 4 MB
    unsigned short* vbuf = kbuf + tok;                   // 4 MB   (~45 MB total)
    float* out = (float*)d_out;

    // pre-convert inputs to bf16 (bandwidth bound)
    TemporalHead_cvt_kernel<<<(int)(XN / 8 + 255) / 256, 256, 0, stream>>>(x, xb, (int)(XN / 8));
    TemporalHead_cvt_kernel<<<(int)(WN / 8 + 255) / 256, 256, 0, stream>>>(Wq, wq, (int)(WN / 8));
    TemporalHead_cvt_kernel<<<(int)(WN / 8 + 255) / 256, 256, 0, stream>>>(Wk, wk, (int)(WN / 8));
    TemporalHead_cvt_kernel<<<(int)(WN / 8 + 255) / 256, 256, 0, stream>>>(Wv, wv, (int)(WN / 8));

    dim3 g1((NBATCH * T_SEQ) / 16, H_DIM / 16);          // 1024 x 8 waves
    TemporalHead_qkv_kernel<<<g1, 32, 0, stream>>>(xb, wq, wk, wv, qb, kbuf, vbuf);

    dim3 g2(T_SEQ / 16, NBATCH);                         // 256 x 4 waves
    TemporalHead_attn_kernel<<<g2, 32, 0, stream>>>(qb, kbuf, vbuf, out);
}
